// TKGCN_V9_22179211117214
// MI455X (gfx1250) — compile-verified, matrix operand order checked by reading, not run on hardware
//
#include <hip/hip_runtime.h>
#include <hip/hip_bf16.h>

// ---------------------------------------------------------------------------
// CDNA5 (gfx1250) implementation: bf16 WMMA GEMMs + f32-atomic edge scatter.
// ---------------------------------------------------------------------------

typedef __bf16 bf16;
typedef __attribute__((ext_vector_type(16))) __bf16 v16bf;
typedef __attribute__((ext_vector_type(8)))  __bf16 v8bf;
typedef __attribute__((ext_vector_type(8)))  float  v8f;

#define NN   50000      // nodes
#define NE   800000     // edges
#define NREL 8
#define NG   64
#define C1   128        // NHID
#define C2   256        // HID2
#define NCLS 10
#define LNE  1e-5f
#define SLP  0.01f

// ---------------- fragment loader --------------------------------------
// 16-bit A/B fragment per ISA 7.12.2: lane (m or n) = lane&15,
// half = lane>>4 : holds K = kb + 8*half .. +7  and  kb + 16 + 8*half .. +7
__device__ __forceinline__ v16bf load_frag(const bf16* __restrict__ p, int half) {
    v8bf lo = *(const v8bf*)(p + 8 * half);
    v8bf hi = *(const v8bf*)(p + 16 + 8 * half);
    v16bf r;
#pragma unroll
    for (int i = 0; i < 8; ++i) { r[i] = lo[i]; r[i + 8] = hi[i]; }
    return r;
}

// ---------------- generic WMMA GEMM ------------------------------------
// C[M,N] = A[M,K] (bf16, row-major) x Bt[N,K]^T (bf16, N-major "transposed B")
// optional bias[N], residual add0[M,N] (f32), leaky-ReLU, dual f32/bf16 out.
// blockIdx.z selects a relation slice of Bt / out (btRel / outRel strides).
__global__ void k_gemm(const bf16* __restrict__ A, const bf16* __restrict__ Bt,
                       const float* __restrict__ bias, const float* __restrict__ add0,
                       float* __restrict__ outF, bf16* __restrict__ outB,
                       int M, int N, int K, int act,
                       size_t btRel, size_t outRel) {
    const int wave  = threadIdx.x >> 5;
    const int lane  = threadIdx.x & 31;
    const int half  = lane >> 4;
    const int l16   = lane & 15;
    const int mBase = blockIdx.x * 16;
    const int nTile = blockIdx.y * 8 + wave;
    if (nTile * 16 >= N) return;           // wave-uniform: EXEC stays all-ones
    const int nBase = nTile * 16;

    const bf16* __restrict__ Arow = A + (size_t)(mBase + l16) * K;
    const bf16* __restrict__ Brow = Bt + btRel * blockIdx.z + (size_t)(nBase + l16) * K;

    v8f acc = {};
    for (int kb = 0; kb < K; kb += 32) {
        __builtin_prefetch(Arow + kb + 64, 0, 1);   // global_prefetch_b8
        v16bf av = load_frag(Arow + kb, half);
        v16bf bv = load_frag(Brow + kb, half);
        acc = __builtin_amdgcn_wmma_f32_16x16x32_bf16(
                  false, av, false, bv, (short)0, acc, false, false);
    }

    const int   n  = nBase + l16;
    const float b0 = bias ? bias[n] : 0.0f;
    const size_t oo = outRel * blockIdx.z;
#pragma unroll
    for (int i = 0; i < 8; ++i) {          // D layout: m = mBase + 8*half + i
        const int    m   = mBase + half * 8 + i;
        const size_t idx = (size_t)m * N + n;
        float v = acc[i] + b0;
        if (add0) v += add0[idx];
        if (act)  v = (v > 0.0f) ? v : v * SLP;
        if (outF) outF[oo + idx] = v;
        if (outB) outB[oo + idx] = (bf16)v;
    }
}

// ---------------- conversion kernels -----------------------------------
__global__ void k_cvt_bf16(const float* __restrict__ s, bf16* __restrict__ d, size_t n) {
    for (size_t i = (size_t)blockIdx.x * blockDim.x + threadIdx.x; i < n;
         i += (size_t)gridDim.x * blockDim.x)
        d[i] = (bf16)s[i];
}
// src: nm x C x D  ->  dst: nm x D x C (bf16)
__global__ void k_cvt_tr(const float* __restrict__ s, bf16* __restrict__ d,
                         int C, int D, int nm) {
    size_t total = (size_t)nm * C * D;
    for (size_t i = (size_t)blockIdx.x * blockDim.x + threadIdx.x; i < total;
         i += (size_t)gridDim.x * blockDim.x) {
        size_t m = i / ((size_t)C * D), r = i % ((size_t)C * D);
        int c = (int)(r / D), dd = (int)(r % D);
        d[m * (size_t)C * D + (size_t)dd * C + c] = (bf16)s[i];
    }
}

// ---------------- edge / graph kernels ---------------------------------
__global__ void k_edge_count(const int* __restrict__ dst, const int* __restrict__ et,
                             float* __restrict__ cnt) {
    int e = blockIdx.x * blockDim.x + threadIdx.x;
    if (e < NE) atomicAdd(&cnt[(size_t)dst[e] * NREL + et[e]], 1.0f);
}
__global__ void k_edge_norm(const int* __restrict__ dst, const int* __restrict__ et,
                            const float* __restrict__ cnt, float* __restrict__ nrm) {
    int e = blockIdx.x * blockDim.x + threadIdx.x;
    if (e < NE) nrm[e] = 1.0f / fmaxf(cnt[(size_t)dst[e] * NREL + et[e]], 1.0f);
}
__global__ void k_node_cnt(const int* __restrict__ batch, float* __restrict__ gc) {
    int n = blockIdx.x * blockDim.x + threadIdx.x;
    if (n < NN) atomicAdd(&gc[batch[n]], 1.0f);
}
// one wave per edge: agg[dst] += norm * xW[etype][src]
__global__ void k_scatter(const int* __restrict__ srcI, const int* __restrict__ dstI,
                          const int* __restrict__ etI, const float* __restrict__ nrm,
                          const bf16* __restrict__ XW, float* __restrict__ agg, int D) {
    int wid  = (blockIdx.x * blockDim.x + threadIdx.x) >> 5;
    int lane = threadIdx.x & 31;
    if (wid >= NE) return;
    int s = srcI[wid], d = dstI[wid], r = etI[wid];
    float nm = nrm[wid];
    const bf16* mp = XW + ((size_t)r * NN + s) * D;
    float*      op = agg + (size_t)d * D;
    for (int c = lane; c < D; c += 32) atomicAdd(op + c, nm * (float)mp[c]);
}
// one wave per node: gsum[batch[n]] += x[n]
__global__ void k_gap_sum(const float* __restrict__ x, const int* __restrict__ batch,
                          float* __restrict__ gsum, int D) {
    int wid  = (blockIdx.x * blockDim.x + threadIdx.x) >> 5;
    int lane = threadIdx.x & 31;
    if (wid >= NN) return;
    const float* p = x + (size_t)wid * D;
    float*       o = gsum + (size_t)batch[wid] * D;
    for (int d = lane; d < D; d += 32) atomicAdd(o + d, p[d]);
}
__global__ void k_gap_mean(const float* __restrict__ gsum, const float* __restrict__ gc,
                           float* __restrict__ gmean, int D) {
    int i = blockIdx.x * blockDim.x + threadIdx.x;
    if (i < NG * D) gmean[i] = gsum[i] / fmaxf(gc[i / D], 1.0f);
}
// cb[n] = concat( hb[n] , gmean[batch[n]] + xf[n] )   (bf16, width 2*Dh)
__global__ void k_concat(const bf16* __restrict__ hb, const float* __restrict__ gmean,
                         const float* __restrict__ xf, const int* __restrict__ batch,
                         bf16* __restrict__ cb, int Dh) {
    size_t i = (size_t)blockIdx.x * blockDim.x + threadIdx.x;
    if (i >= (size_t)NN * Dh) return;
    int n = (int)(i / Dh), d = (int)(i % Dh);
    int g = batch[n];
    cb[(size_t)n * 2 * Dh + d]      = hb[(size_t)n * Dh + d];
    cb[(size_t)n * 2 * Dh + Dh + d] =
        (bf16)(gmean[(size_t)g * Dh + d] + xf[(size_t)n * Dh + d]);
}
// LayerNorm, one wave per row of width D (=256)
__global__ void k_ln(const float* __restrict__ in, const float* __restrict__ g,
                     const float* __restrict__ b, float* __restrict__ outF,
                     bf16* __restrict__ outB, int D) {
    int row  = (blockIdx.x * blockDim.x + threadIdx.x) >> 5;
    int lane = threadIdx.x & 31;
    if (row >= NN) return;
    const float* p = in + (size_t)row * D;
    float s = 0.f, s2 = 0.f;
    for (int d = lane; d < D; d += 32) { float v = p[d]; s += v; s2 += v * v; }
#pragma unroll
    for (int m = 16; m >= 1; m >>= 1) { s += __shfl_xor(s, m, 32); s2 += __shfl_xor(s2, m, 32); }
    float mean = s / D;
    float var  = s2 / D - mean * mean;
    float rstd = rsqrtf(var + LNE);
    for (int d = lane; d < D; d += 32) {
        float v = (p[d] - mean) * rstd * g[d] + b[d];
        outF[(size_t)row * D + d] = v;
        if (outB) outB[(size_t)row * D + d] = (bf16)v;
    }
}
// out[g,c] = (gsum[g]/cnt[g]) . out_w[c] + out_b[c]     (64 x 10, tiny)
__global__ void k_final(const float* __restrict__ gsum, const float* __restrict__ gc,
                        const float* __restrict__ w, const float* __restrict__ bo,
                        float* __restrict__ out) {
    int i = blockIdx.x * blockDim.x + threadIdx.x;
    if (i >= NG * NCLS) return;
    int g = i / NCLS, c = i % NCLS;
    float inv = 1.0f / fmaxf(gc[g], 1.0f);
    float s = 0.f;
    for (int k = 0; k < C2; ++k)
        s += gsum[(size_t)g * C2 + k] * inv * w[(size_t)c * C2 + k];
    out[i] = s + bo[c];
}

// ---------------------------------------------------------------------------
extern "C" void kernel_launch(void* const* d_in, const int* in_sizes, int n_in,
                              void* d_out, int out_size, void* d_ws, size_t ws_size,
                              hipStream_t stream) {
    (void)in_sizes; (void)n_in; (void)out_size; (void)ws_size;

    const float* x     = (const float*)d_in[0];
    const int*   ei    = (const int*)d_in[1];
    const int*   srcI  = ei;
    const int*   dstI  = ei + NE;
    const int*   etI   = (const int*)d_in[2];
    const int*   batch = (const int*)d_in[3];
    const float* Ws    = (const float*)d_in[4];
    const float* Rs    = (const float*)d_in[5];
    const float* Bs    = (const float*)d_in[6];
    const float* Wf1   = (const float*)d_in[7];
    const float* Rf1   = (const float*)d_in[8];
    const float* Bf1   = (const float*)d_in[9];
    const float* Wf23  = (const float*)d_in[10];
    const float* Rf23  = (const float*)d_in[11];
    const float* Bf23  = (const float*)d_in[12];
    const float* Wf4   = (const float*)d_in[13];
    const float* Rf4   = (const float*)d_in[14];
    const float* Bf4   = (const float*)d_in[15];
    const float* AinW  = (const float*)d_in[16];
    const float* AinB  = (const float*)d_in[17];
    const float* AoutW = (const float*)d_in[18];
    const float* AoutB = (const float*)d_in[19];
    const float* FcW   = (const float*)d_in[20];
    const float* FcB   = (const float*)d_in[21];
    const float* LnG   = (const float*)d_in[22];
    const float* LnB   = (const float*)d_in[23];
    const float* OutW  = (const float*)d_in[24];
    const float* OutB  = (const float*)d_in[25];
    float* out = (float*)d_out;

    // ---- bump allocator over d_ws ----
    char* wp = (char*)d_ws;
    auto carve = [&](size_t bytes) -> char* {
        char* r = wp; wp += (bytes + 255) & ~(size_t)255; return r;
    };
    float* normE = (float*)carve((size_t)NE * 4);
    float* cntNR = (float*)carve((size_t)NN * NREL * 4);
    float* gcnt  = (float*)carve(NG * 4);
    float* gsum  = (float*)carve((size_t)NG * C2 * 4);
    float* gmean = (float*)carve((size_t)NG * C2 * 4);
    bf16*  WsT   = (bf16*)carve((size_t)3 * NREL * C1 * C1 * 2);
    bf16*  RsT   = (bf16*)carve((size_t)3 * C1 * C1 * 2);
    bf16*  Wf1T  = (bf16*)carve((size_t)NREL * C1 * C1 * 2);
    bf16*  Rf1T  = (bf16*)carve((size_t)C1 * C1 * 2);
    bf16*  Wf23T = (bf16*)carve((size_t)2 * NREL * C1 * C2 * 2);
    bf16*  Rf23T = (bf16*)carve((size_t)2 * C1 * C2 * 2);
    bf16*  Wf4T  = (bf16*)carve((size_t)NREL * C2 * C2 * 2);
    bf16*  Rf4T  = (bf16*)carve((size_t)C2 * C2 * 2);
    bf16*  WvB   = (bf16*)carve((size_t)C2 * C2 * 2);
    bf16*  WoB   = (bf16*)carve((size_t)C2 * C2 * 2);
    bf16*  WfcB  = (bf16*)carve((size_t)C2 * C2 * 2);
    bf16*  x0b   = (bf16*)carve((size_t)NN * C1 * 2);
    bf16*  Ab0   = (bf16*)carve((size_t)NN * C2 * 2);
    bf16*  Ab1   = (bf16*)carve((size_t)NN * C2 * 2);
    float* x1f   = (float*)carve((size_t)NN * C1 * 4);
    float* x2f   = (float*)carve((size_t)NN * C1 * 4);
    float* x3f   = (float*)carve((size_t)NN * C1 * 4);
    bf16*  XWB   = (bf16*)carve((size_t)NREL * NN * C2 * 2); // 205 MB, reused late
    float* AGG   = (float*)carve((size_t)NN * C2 * 4);       // also hosts a0
    float* qf    = (float*)carve((size_t)NN * C2 * 4);
    float* af    = (float*)carve((size_t)NN * C2 * 4);
    float* ff    = (float*)XWB;                 // XWB dead after last scatter
    float* off_  = (float*)XWB + (size_t)NN * C2;

    const int T = 256;
    auto gb = [](size_t n, int t) { return (int)((n + t - 1) / t); };

    auto gemm = [&](const bf16* A, const bf16* Bt, const float* bias, const float* add0,
                    float* outF, bf16* outB, int N, int K, int Z,
                    size_t btRel, size_t outRel, int act) {
        dim3 g(NN / 16, (N / 16 + 7) / 8, Z);
        k_gemm<<<g, 256, 0, stream>>>(A, Bt, bias, add0, outF, outB, NN, N, K, act,
                                      btRel, outRel);
    };
    auto rgcn = [&](const bf16* Ain, int Cin, int Dout, const bf16* WT, const bf16* RT,
                    const float* bias, float* outF, bf16* outB) {
        gemm(Ain, WT, nullptr, nullptr, nullptr, XWB, Dout, Cin, NREL,
             (size_t)Dout * Cin, (size_t)NN * Dout, 0);                 // xW per rel
        hipMemsetAsync(AGG, 0, (size_t)NN * Dout * 4, stream);
        k_scatter<<<gb((size_t)NE * 32, T), T, 0, stream>>>(srcI, dstI, etI, normE,
                                                            XWB, AGG, Dout);
        gemm(Ain, RT, bias, AGG, outF, outB, Dout, Cin, 1, 0, 0, 1);    // +root+act
    };
    auto gap = [&](const float* xf, int D) {
        hipMemsetAsync(gsum, 0, (size_t)NG * C2 * 4, stream);
        k_gap_sum<<<gb((size_t)NN * 32, T), T, 0, stream>>>(xf, batch, gsum, D);
        k_gap_mean<<<gb((size_t)NG * D, T), T, 0, stream>>>(gsum, gcnt, gmean, D);
    };

    // ---- one-time prep: norms, graph counts, bf16 weights, bf16 x0 ----
    hipMemsetAsync(cntNR, 0, (size_t)NN * NREL * 4, stream);
    hipMemsetAsync(gcnt, 0, NG * 4, stream);
    k_edge_count<<<gb(NE, T), T, 0, stream>>>(dstI, etI, cntNR);
    k_edge_norm<<<gb(NE, T), T, 0, stream>>>(dstI, etI, cntNR, normE);
    k_node_cnt<<<gb(NN, T), T, 0, stream>>>(batch, gcnt);
    k_cvt_tr<<<2048, T, 0, stream>>>(Ws, WsT, C1, C1, 3 * NREL);
    k_cvt_tr<<<512, T, 0, stream>>>(Rs, RsT, C1, C1, 3);
    k_cvt_tr<<<1024, T, 0, stream>>>(Wf1, Wf1T, C1, C1, NREL);
    k_cvt_tr<<<256, T, 0, stream>>>(Rf1, Rf1T, C1, C1, 1);
    k_cvt_tr<<<2048, T, 0, stream>>>(Wf23, Wf23T, C2, C1, 2 * NREL);
    k_cvt_tr<<<512, T, 0, stream>>>(Rf23, Rf23T, C2, C1, 2);
    k_cvt_tr<<<2048, T, 0, stream>>>(Wf4, Wf4T, C2, C2, NREL);
    k_cvt_tr<<<512, T, 0, stream>>>(Rf4, Rf4T, C2, C2, 1);
    k_cvt_bf16<<<512, T, 0, stream>>>(AinW + (size_t)2 * C2 * C2, WvB, (size_t)C2 * C2);
    k_cvt_bf16<<<512, T, 0, stream>>>(AoutW, WoB, (size_t)C2 * C2);
    k_cvt_bf16<<<512, T, 0, stream>>>(FcW, WfcB, (size_t)C2 * C2);
    k_cvt_bf16<<<4096, T, 0, stream>>>(x, x0b, (size_t)NN * C1);

    // ---- GCN trunk ----
    rgcn(x0b, C1, C1, WsT, RsT, Bs, x1f, Ab0);                                  // x1
    rgcn(Ab0, C1, C1, WsT + (size_t)NREL * C1 * C1, RsT + (size_t)C1 * C1,
         Bs + C1, x2f, Ab1);                                                    // x2
    rgcn(Ab1, C1, C1, WsT + (size_t)2 * NREL * C1 * C1, RsT + (size_t)2 * C1 * C1,
         Bs + 2 * C1, x3f, nullptr);                                            // x3
    rgcn(x0b, C1, C1, Wf1T, Rf1T, Bf1, nullptr, Ab1);                           // h1

    gap(x1f, C1);
    k_concat<<<gb((size_t)NN * C1, T), T, 0, stream>>>(Ab1, gmean, x1f, batch, Ab0, C1);
    rgcn(Ab0, C2, C1, Wf23T, Rf23T, Bf23, nullptr, Ab1);                        // h2

    gap(x2f, C1);
    k_concat<<<gb((size_t)NN * C1, T), T, 0, stream>>>(Ab1, gmean, x2f, batch, Ab0, C1);
    rgcn(Ab0, C2, C1, Wf23T + (size_t)NREL * C1 * C2, Rf23T + (size_t)C1 * C2,
         Bf23 + C1, nullptr, Ab1);                                              // h3

    gap(x3f, C1);
    k_concat<<<gb((size_t)NN * C1, T), T, 0, stream>>>(Ab1, gmean, x3f, batch, Ab0, C1);
    rgcn(Ab0, C2, C2, Wf4T, Rf4T, Bf4, qf, Ab1);                                // q

    // ---- degenerate MHA (L=1 => softmax==1 => o = v) + encoder ----
    gemm(Ab1, WvB, AinB + 2 * C2, nullptr, nullptr, Ab0, C2, C2, 1, 0, 0, 0);   // v
    gemm(Ab0, WoB, AoutB, qf, AGG, nullptr, C2, C2, 1, 0, 0, 0);                // a0+Q
    k_ln<<<gb((size_t)NN * 32, T), T, 0, stream>>>(AGG, LnG, LnB, af, Ab1, C2); // a
    gemm(Ab1, WfcB, FcB, af, ff, nullptr, C2, C2, 1, 0, 0, 0);                  // f+a
    k_ln<<<gb((size_t)NN * 32, T), T, 0, stream>>>(ff, LnG, LnB, off_, nullptr, C2);

    // ---- pooled classifier ----
    hipMemsetAsync(gsum, 0, (size_t)NG * C2 * 4, stream);
    k_gap_sum<<<gb((size_t)NN * 32, T), T, 0, stream>>>(off_, batch, gsum, C2);
    k_final<<<gb(NG * NCLS, T), T, 0, stream>>>(gsum, gcnt, OutW, OutB, out);
}
